// GraphSAGE_23476291240659
// MI455X (gfx1250) — matrix operand order repaired and checked by reading, not compiled, not run on hardware
//
#include <hip/hip_runtime.h>
#include <hip/hip_bf16.h>

#define N_NODES 50000
#define N_EDGES 800000
#define N_GRAPHS 500
#define H 90
#define HP 96      // N-dim padded to 6x16
#define KP 192     // K-dim (2*H=180) padded to 6x32

typedef __attribute__((ext_vector_type(16))) _Float16 v16h;
typedef __attribute__((ext_vector_type(8)))  _Float16 v8h;
typedef __attribute__((ext_vector_type(8)))  float    v8f;
typedef __attribute__((ext_vector_type(4)))  unsigned int v4u;
typedef __attribute__((ext_vector_type(8)))  int      v8i;
typedef __attribute__((ext_vector_type(4)))  int      v4i;

union FragH { v16h v; v8h half[2]; };

// ---------------- utility kernels ----------------

__global__ void zero_kernel(float* __restrict__ p, int n) {
    int i = blockIdx.x * blockDim.x + threadIdx.x;
    if (i < n) p[i] = 0.0f;
}

__global__ void count_edges_kernel(const int* __restrict__ dst, float* __restrict__ cnt) {
    int e = blockIdx.x * blockDim.x + threadIdx.x;
    if (e < N_EDGES) atomicAdd(&cnt[dst[e]], 1.0f);
}

// one thread per (edge, feature): coalesced over features, f32 global atomics (L2-resident)
__global__ void scatter_edges_kernel(const float* __restrict__ xin,
                                     const int* __restrict__ src,
                                     const int* __restrict__ dst,
                                     float* __restrict__ aggr) {
    int idx = blockIdx.x * blockDim.x + threadIdx.x;   // < E*H = 72e6
    if (idx >= N_EDGES * H) return;
    int e = idx / H;
    int h = idx - e * H;
    atomicAdd(&aggr[dst[e] * H + h], xin[src[e] * H + h]);
}

// ---------------- f16 operand packing ----------------
// xcat[n][k] = k<H ? aggr[n][k]/max(cnt,1) : (k<2H ? x[n][k-H] : 0)   [N][192] f16
__global__ void pack_x_kernel(const float* __restrict__ aggr, const float* __restrict__ cnt,
                              const float* __restrict__ xin, _Float16* __restrict__ xcat) {
    int idx = blockIdx.x * blockDim.x + threadIdx.x;   // < N*KP
    if (idx >= N_NODES * KP) return;
    int n = idx / KP, k = idx - n * KP;
    float v = 0.0f;
    if (k < H) {
        float c = cnt[n]; c = c > 1.0f ? c : 1.0f;
        v = aggr[n * H + k] / c;
    } else if (k < 2 * H) {
        v = xin[n * H + (k - H)];
    }
    xcat[idx] = (_Float16)v;
}

// wcat[j][k] = k<H ? Wl[j][k] : (k<2H ? Wr[j][k-H] : 0)   [96][192] f16
__global__ void pack_w_kernel(const float* __restrict__ Wl, const float* __restrict__ Wr,
                              _Float16* __restrict__ wcat) {
    int idx = blockIdx.x * blockDim.x + threadIdx.x;   // < HP*KP
    if (idx >= HP * KP) return;
    int j = idx / KP, k = idx - j * KP;
    float v = 0.0f;
    if (j < H) {
        if (k < H)          v = Wl[j * H + k];
        else if (k < 2 * H) v = Wr[j * H + (k - H)];
    }
    wcat[idx] = (_Float16)v;
}

// ---------------- TDM 2D tile load: global(f16, row stride dim0) -> LDS ----------------
// D# per CDNA5 ISA ch.8: group0 = {count, lds_addr, global_addr, type=2},
// group1 = {data_size=2B, tensor_dim0/1, tile_dim0/1, dim0_stride}. Groups 2/3 zero
// (tile_dim2 = tile_dim3 = 0 -> unused; 2D tensor). OOB rows (n >= tensor_dim1) read zero.
__device__ __forceinline__ void tdm_load_2d(const void* gptr, void* ldsptr,
                                            int dim0, int dim1, int tile0, int tile1) {
    unsigned long long ga = (unsigned long long)(uintptr_t)gptr;
    unsigned int la = (unsigned int)(uintptr_t)ldsptr;   // flat addr[31:0] == LDS byte offset
    v4u g0;
    g0[0] = 1u;                                          // count=1, user descriptor
    g0[1] = la;                                          // lds_addr [63:32]
    g0[2] = (unsigned int)ga;                            // global_addr[31:0]  [95:64]
    g0[3] = (unsigned int)((ga >> 32) & 0x01FFFFFFu)     // global_addr[56:32] [120:96]
          | (2u << 30);                                  // type=2 ("image")   [127:126]
    v8i g1;
    g1[0] = (int)(1u << 16);                             // data_size=1 (2 bytes); mask=0
    g1[1] = (int)(((unsigned)dim0 & 0xFFFFu) << 16);     // tensor_dim0[15:0] @ [63:48]
    g1[2] = (int)((((unsigned)dim0 >> 16) & 0xFFFFu)     // tensor_dim0[31:16]
          | (((unsigned)dim1 & 0xFFFFu) << 16));         // tensor_dim1[15:0]
    g1[3] = (int)((((unsigned)dim1 >> 16) & 0xFFFFu)     // tensor_dim1[31:16]
          | (((unsigned)tile0 & 0xFFFFu) << 16));        // tile_dim0 @ [127:112]
    g1[4] = (int)((unsigned)tile1 & 0xFFFFu);            // tile_dim1; tile_dim2=0
    g1[5] = dim0;                                        // tensor_dim0_stride[31:0]
    g1[6] = 0;                                           // stride0 hi / dim1_stride lo
    g1[7] = 0;
    v4i z4 = {0, 0, 0, 0};
    v8i z8 = {0, 0, 0, 0, 0, 0, 0, 0};
    __builtin_amdgcn_tensor_load_to_lds(g0, g1, z4, z4, z8, 0);
}

// ---------------- fused SAGE GEMM (TDM staging + WMMA) ----------------
// out[n,:] = relu( xcat[n,:] @ wcat^T + bl )  with K=192, N-dim 96 (store j<90 only)
// Block: 128 threads = 4 waves; M-tile 64 rows; each wave: 16 rows x 96 cols.
__global__ void __launch_bounds__(128)
sage_gemm_kernel(const _Float16* __restrict__ xcat, const _Float16* __restrict__ wcat,
                 const float* __restrict__ bl, float* __restrict__ out) {
    __shared__ _Float16 sW[HP * KP];   // 36 KB
    __shared__ _Float16 sX[64 * KP];   // 24 KB

    const int tid = threadIdx.x;
    const int blockRow = blockIdx.x * 64;

    if (tid < 32) {   // wave 0 drives the Tensor Data Mover (uniform operands -> SGPRs)
        tdm_load_2d(wcat, sW, KP, HP, KP, HP);
        tdm_load_2d(xcat + (size_t)blockRow * KP, sX, KP, N_NODES - blockRow, KP, 64);
        __builtin_amdgcn_s_wait_tensorcnt(0);
    }
    __syncthreads();

    const int wave = tid >> 5;
    const int lane = tid & 31;
    const int lm   = lane & 15;   // M (for A) or N (for B) within the 16x16 tile
    const int lh   = lane >> 4;   // K-half selector per ISA fragment layout

    v8f acc[6] = {};              // 6 N-subtiles of f32 16x16 accumulators

    const _Float16* aRow = sX + (wave * 16 + lm) * KP;
    for (int kk = 0; kk < 6; ++kk) {
        const int K0 = kk * 32;
        FragH a;
        // A 16x32 f16 frag: VGPR0-3 <- K = lh*8+[0..7], VGPR4-7 <- K = 16+lh*8+[0..7]
        a.half[0] = *(const v8h*)(aRow + K0 + lh * 8);
        a.half[1] = *(const v8h*)(aRow + K0 + 16 + lh * 8);
#pragma unroll
        for (int nt = 0; nt < 6; ++nt) {
            // B 32x16 frag of wcat^T: lane holds column j = nt*16+lm,
            // K = K0 + lh*8 + [0..7], K0 + 16 + lh*8 + [0..7] in sW row j
            const _Float16* bRow = sW + (nt * 16 + lm) * KP + K0 + lh * 8;
            FragH b;
            b.half[0] = *(const v8h*)(bRow);
            b.half[1] = *(const v8h*)(bRow + 16);
            acc[nt] = __builtin_amdgcn_wmma_f32_16x16x32_f16(
                false, a.v, false, b.v, (short)0, acc[nt], false, false);
        }
    }

    // D layout: VGPR r -> M = r + 8*lh, N = lm
#pragma unroll
    for (int nt = 0; nt < 6; ++nt) {
        int j = nt * 16 + lm;
        if (j >= H) continue;
        float bb = bl[j];
#pragma unroll
        for (int r = 0; r < 8; ++r) {
            int row = blockRow + wave * 16 + r + 8 * lh;
            if (row < N_NODES) {
                float v = acc[nt][r] + bb;
                out[row * H + j] = v > 0.0f ? v : 0.0f;
            }
        }
    }
}

// ---------------- pooling + MLP tail ----------------

__global__ void pool_sum_kernel(const float* __restrict__ h3, const int* __restrict__ batch,
                                float* __restrict__ gs) {
    int idx = blockIdx.x * blockDim.x + threadIdx.x;   // < N*H
    if (idx >= N_NODES * H) return;
    int n = idx / H;
    int h = idx - n * H;
    atomicAdd(&gs[batch[n] * H + h], h3[idx]);
}

__global__ void pool_cnt_kernel(const int* __restrict__ batch, float* __restrict__ gc) {
    int n = blockIdx.x * blockDim.x + threadIdx.x;
    if (n < N_NODES) atomicAdd(&gc[batch[n]], 1.0f);
}

__global__ void mlp_kernel(const float* __restrict__ gs, const float* __restrict__ gc,
                           const float* __restrict__ fc1_w, const float* __restrict__ fc1_b,
                           const float* __restrict__ fc2_w, const float* __restrict__ fc2_b,
                           float* __restrict__ outp) {
    int g = blockIdx.x * blockDim.x + threadIdx.x;
    if (g >= N_GRAPHS) return;
    float c = gc[g]; c = c > 1.0f ? c : 1.0f;
    float inv = 1.0f / c;
    float acc = fc2_b[0];
    for (int o = 0; o < 32; ++o) {
        float s = fc1_b[o];
        for (int h = 0; h < H; ++h)
            s += fc1_w[o * H + h] * (gs[g * H + h] * inv);
        s = s > 0.0f ? s : 0.0f;
        acc += fc2_w[o] * s;
    }
    outp[g] = acc;
}

// ---------------- launch ----------------

extern "C" void kernel_launch(void* const* d_in, const int* in_sizes, int n_in,
                              void* d_out, int out_size, void* d_ws, size_t ws_size,
                              hipStream_t stream) {
    const float* x     = (const float*)d_in[0];
    const int*   eidx  = (const int*)d_in[1];
    const int*   batch = (const int*)d_in[2];
    const float* W1l = (const float*)d_in[3];
    const float* b1l = (const float*)d_in[4];
    const float* W1r = (const float*)d_in[5];
    const float* W2l = (const float*)d_in[6];
    const float* b2l = (const float*)d_in[7];
    const float* W2r = (const float*)d_in[8];
    const float* W3l = (const float*)d_in[9];
    const float* b3l = (const float*)d_in[10];
    const float* W3r = (const float*)d_in[11];
    const float* fc1_w = (const float*)d_in[12];
    const float* fc1_b = (const float*)d_in[13];
    const float* fc2_w = (const float*)d_in[14];
    const float* fc2_b = (const float*)d_in[15];
    float* out = (float*)d_out;

    const int* src = eidx;
    const int* dst = eidx + N_EDGES;

    // workspace layout
    float* ws   = (float*)d_ws;
    float* hA   = ws;                       // N*H f32
    float* hB   = hA + N_NODES * H;         // N*H f32
    float* aggr = hB + N_NODES * H;         // N*H f32
    float* cnt  = aggr + N_NODES * H;       // N f32
    float* gs   = cnt + N_NODES;            // G*H f32
    float* gc   = gs + N_GRAPHS * H;        // G f32
    _Float16* xcat = (_Float16*)(gc + N_GRAPHS);            // N*KP f16
    _Float16* wcat = (_Float16*)(xcat + N_NODES * KP);      // HP*KP f16

    const int NH   = N_NODES * H;
    const int EH   = N_EDGES * H;
    const int NKP  = N_NODES * KP;
    const int ZB   = (NH + 255) / 256;
    const int EB   = (N_EDGES + 255) / 256;
    const int EHB  = (EH + 255) / 256;
    const int PXB  = (NKP + 255) / 256;
    const int PWB  = (HP * KP + 255) / 256;
    const int GMMB = (N_NODES + 63) / 64;

    // init accumulators
    zero_kernel<<<ZB, 256, 0, stream>>>(aggr, NH);
    zero_kernel<<<(N_NODES + 255) / 256, 256, 0, stream>>>(cnt, N_NODES);
    zero_kernel<<<(N_GRAPHS * H + 255) / 256, 256, 0, stream>>>(gs, N_GRAPHS * H);
    zero_kernel<<<(N_GRAPHS + 255) / 256, 256, 0, stream>>>(gc, N_GRAPHS);

    // in-degree counts (shared by all 3 layers)
    count_edges_kernel<<<EB, 256, 0, stream>>>(dst, cnt);

    // layer 1
    scatter_edges_kernel<<<EHB, 256, 0, stream>>>(x, src, dst, aggr);
    pack_x_kernel<<<PXB, 256, 0, stream>>>(aggr, cnt, x, xcat);
    pack_w_kernel<<<PWB, 256, 0, stream>>>(W1l, W1r, wcat);
    sage_gemm_kernel<<<GMMB, 128, 0, stream>>>(xcat, wcat, b1l, hA);

    // layer 2
    zero_kernel<<<ZB, 256, 0, stream>>>(aggr, NH);
    scatter_edges_kernel<<<EHB, 256, 0, stream>>>(hA, src, dst, aggr);
    pack_x_kernel<<<PXB, 256, 0, stream>>>(aggr, cnt, hA, xcat);
    pack_w_kernel<<<PWB, 256, 0, stream>>>(W2l, W2r, wcat);
    sage_gemm_kernel<<<GMMB, 128, 0, stream>>>(xcat, wcat, b2l, hB);

    // layer 3
    zero_kernel<<<ZB, 256, 0, stream>>>(aggr, NH);
    scatter_edges_kernel<<<EHB, 256, 0, stream>>>(hB, src, dst, aggr);
    pack_x_kernel<<<PXB, 256, 0, stream>>>(aggr, cnt, hB, xcat);
    pack_w_kernel<<<PWB, 256, 0, stream>>>(W3l, W3r, wcat);
    sage_gemm_kernel<<<GMMB, 128, 0, stream>>>(xcat, wcat, b3l, hA);

    // global mean pool + MLP head
    pool_sum_kernel<<<(NH + 255) / 256, 256, 0, stream>>>(hA, batch, gs);
    pool_cnt_kernel<<<(N_NODES + 255) / 256, 256, 0, stream>>>(batch, gc);
    mlp_kernel<<<(N_GRAPHS + 255) / 256, 256, 0, stream>>>(gs, gc, fc1_w, fc1_b,
                                                           fc2_w, fc2_b, out);
}
